// HSTU_BSA_Triton_23201413333344
// MI455X (gfx1250) — matrix operand order repaired and checked by reading, not compiled, not run on hardware
//
#include <hip/hip_runtime.h>
#include <hip/hip_bf16.h>

// ---------------- problem constants (from reference) ----------------
#define B_    4
#define T_    2048
#define H_    4
#define D_    64
#define BS_   32
#define NB_   64          // T_/BS_
#define S_    4
#define SCALE 0.125f      // 1/sqrt(64)

typedef __attribute__((ext_vector_type(16))) __bf16       v16bf;
typedef __attribute__((ext_vector_type(8)))  float        v8f;
typedef __attribute__((ext_vector_type(4)))  float        f32x4;
typedef __attribute__((ext_vector_type(2)))  float        f32x2;
typedef __attribute__((ext_vector_type(4)))  unsigned int u32x4;

union AFrag { v16bf v; u32x4 q[2]; };

// =====================================================================
// Kernel 1: mean-pool K,V over 32-token blocks.
//   kc  : bf16 [B][H][Nb][D]   (row n contiguous == WMMA B-frag for Q·Kcmp^T)
//   vcT : bf16 [B][H][D][Nb]   (row d contiguous == WMMA B-frag for P·Vcmp)
// grid = B*Nb blocks, 256 threads = (h,d) pairs.
// =====================================================================
__global__ __launch_bounds__(256) void pool_kernel(
    const float* __restrict__ k, const float* __restrict__ v,
    __bf16* __restrict__ kc, __bf16* __restrict__ vcT) {
  const int b  = blockIdx.x / NB_;
  const int nb = blockIdx.x % NB_;
  const int h  = threadIdx.x >> 6;
  const int d  = threadIdx.x & 63;
  const size_t base = ((size_t)(b * T_ + nb * BS_) * H_ + h) * D_ + d;
  float ks = 0.0f, vs = 0.0f;
#pragma unroll 8
  for (int j = 0; j < BS_; ++j) {
    ks += k[base + (size_t)j * H_ * D_];
    vs += v[base + (size_t)j * H_ * D_];
  }
  ks *= (1.0f / BS_);
  vs *= (1.0f / BS_);
  kc [((size_t)(b * H_ + h) * NB_ + nb) * D_ + d ] = (__bf16)ks;
  vcT[((size_t)(b * H_ + h) * D_  + d ) * NB_ + nb] = (__bf16)vs;
}

// =====================================================================
// Kernel 2: compressed attention + top-k block selection.
// One wave = one 16-row Q tile of a fixed (b,h). 4 waves / block,
// all 4 waves share one (b,h): Kcmp/VcmpT are async-copied to LDS once
// per block (global_load_async_to_lds_b128, ASYNCcnt) and all WMMA
// B-fragments are fed from LDS.
// =====================================================================
__global__ __launch_bounds__(128) void cmp_attn_kernel(
    const float* __restrict__ q, const float* __restrict__ g_cmp,
    const __bf16* __restrict__ kc, const __bf16* __restrict__ vcT,
    int* __restrict__ topk, float* __restrict__ out) {
  // padded strides: sc 68 f32 (bank spread), pl/kv 72|80 bf16 (16B multiple)
  __shared__ __align__(16) float  sc_s[4][16][68];
  __shared__ __align__(16) __bf16 pl_s[4][16][80];
  __shared__ __align__(16) __bf16 kc_lds[NB_][72];   // [block n][dim d]
  __shared__ __align__(16) __bf16 vc_lds[D_][72];    // [dim d][block n]

  const int tid  = threadIdx.x;
  const int w    = tid >> 5;
  const int lane = tid & 31;
  const int row  = lane & 15;     // A row / B column / D column
  const int hi   = lane >> 4;     // half-wave select

  const int gtile = blockIdx.x * 4 + w;     // 0..2047
  const int bh    = gtile >> 7;             // /128 tiles per (b,h)
  const int tile  = gtile & 127;
  const int b     = bh >> 2;
  const int h     = bh & 3;
  const int t0    = tile * 16;

  // ---- async-stage Kcmp and VcmpT for this (b,h) into LDS ----
  // 64 rows x 64 bf16 = 512 16B-chunks per array; 128 threads x 4 chunks.
  {
    const __bf16* kcb = kc  + (size_t)(b * H_ + h) * NB_ * D_;
    const __bf16* vcb = vcT + (size_t)(b * H_ + h) * D_ * NB_;
#pragma unroll
    for (int c = 0; c < 4; ++c) {
      const int chunk = c * 128 + tid;     // 0..511
      const int r  = chunk >> 3;           // row 0..63
      const int cc = (chunk & 7) * 8;      // element offset in row
      const unsigned lk = (unsigned)(uintptr_t)&kc_lds[r][cc];
      const unsigned lv = (unsigned)(uintptr_t)&vc_lds[r][cc];
      const __bf16* gk = kcb + r * D_  + cc;
      const __bf16* gv = vcb + r * NB_ + cc;
      asm volatile("global_load_async_to_lds_b128 %0, %1, off"
                   :: "v"(lk), "v"(gk) : "memory");
      asm volatile("global_load_async_to_lds_b128 %0, %1, off"
                   :: "v"(lv), "v"(gv) : "memory");
    }
  }

  // ---- A-fragments of Q (f32 -> bf16 on the fly), 2 K-steps of 32 ----
  AFrag aQ[2];
  {
    const float* qrow = q + ((size_t)(b * T_ + t0 + row) * H_ + h) * D_;
#pragma unroll
    for (int ks = 0; ks < 2; ++ks) {
      const int c1 = ks * 32 + hi * 8;       // K k0+hi*8 .. +7
      const int c2 = ks * 32 + 16 + hi * 8;  // K k0+16+hi*8 .. +7
#pragma unroll
      for (int e = 0; e < 8; ++e) aQ[ks].v[e]     = (__bf16)qrow[c1 + e];
#pragma unroll
      for (int e = 0; e < 8; ++e) aQ[ks].v[e + 8] = (__bf16)qrow[c2 + e];
    }
  }

  // wait for this wave's async copies; barrier makes them block-visible
  asm volatile("s_wait_asynccnt 0x0" ::: "memory");
  __syncthreads();

  // ---- scores = Q * Kcmp^T : 4 N-tiles x 2 K-steps of WMMA bf16 ----
  v8f accS[4];
#pragma unroll
  for (int nt = 0; nt < 4; ++nt) {
    v8f c = {};
#pragma unroll
    for (int ks = 0; ks < 2; ++ks) {
      const __bf16* bp = &kc_lds[nt * 16 + row][ks * 32 + hi * 8];
      AFrag bK;
      bK.q[0] = *(const u32x4*)(bp);
      bK.q[1] = *(const u32x4*)(bp + 16);
      c = __builtin_amdgcn_wmma_f32_16x16x32_bf16(false, aQ[ks].v, false, bK.v,
                                                  (short)0, c, false, false);
    }
    accS[nt] = c;
  }

  // ---- spill raw scores to LDS (C/D layout: M = hi*8+r, N = nt*16+row) ----
#pragma unroll
  for (int nt = 0; nt < 4; ++nt)
#pragma unroll
    for (int r = 0; r < 8; ++r)
      sc_s[w][hi * 8 + r][nt * 16 + row] = accS[nt][r];
  asm volatile("s_wait_dscnt 0x0" ::: "memory");

  // ---- per-row top-4 valid blocks (lanes 0..15 own one row each) ----
  if (lane < 16) {
    const int t    = t0 + lane;
    const int nmax = t >> 5;                    // last valid compressed block
    unsigned long long chosen = 0ull;
    int* tp = topk + ((size_t)(b * H_ + h) * T_ + t) * S_;
    for (int s = 0; s < S_; ++s) {
      float best = -3.0e38f;
      int   bi   = -1;
      for (int n = 0; n <= nmax; ++n) {
        if (!((chosen >> n) & 1ull)) {
          const float sv = sc_s[w][lane][n];
          if (sv > best) { best = sv; bi = n; }
        }
      }
      if (bi >= 0) chosen |= (1ull << bi);
      tp[s] = bi;
    }
  }

  // ---- P = silu(scale*score) * causal-block mask, stored bf16 ----
#pragma unroll
  for (int j = 0; j < 32; ++j) {
    const int M    = j >> 1;
    const int N    = ((j & 1) << 5) + lane;
    const int nmax = (t0 + M) >> 5;
    const float s  = sc_s[w][M][N] * SCALE;
    const float p  = s / (1.0f + __expf(-s));
    pl_s[w][M][N]  = (__bf16)((N <= nmax) ? p : 0.0f);
  }
  asm volatile("s_wait_dscnt 0x0" ::: "memory");

  // ---- O = P * Vcmp : A-frags and B-frags both from LDS ----
  AFrag aP[2];
#pragma unroll
  for (int ks = 0; ks < 2; ++ks) {
    const __bf16* pr = &pl_s[w][row][0];
    aP[ks].q[0] = *(const u32x4*)(pr + ks * 32 + hi * 8);
    aP[ks].q[1] = *(const u32x4*)(pr + ks * 32 + 16 + hi * 8);
  }
  v8f accO[4];
#pragma unroll
  for (int dt = 0; dt < 4; ++dt) {
    v8f c = {};
#pragma unroll
    for (int ks = 0; ks < 2; ++ks) {
      const __bf16* bp = &vc_lds[dt * 16 + row][ks * 32 + hi * 8];
      AFrag bV;
      bV.q[0] = *(const u32x4*)(bp);
      bV.q[1] = *(const u32x4*)(bp + 16);
      c = __builtin_amdgcn_wmma_f32_16x16x32_bf16(false, aP[ks].v, false, bV.v,
                                                  (short)0, c, false, false);
    }
    accO[dt] = c;
  }

  // ---- write o_cmp * g_cmp ----
#pragma unroll
  for (int r = 0; r < 8; ++r) {
    const int t   = t0 + hi * 8 + r;
    const float gc = g_cmp[(size_t)(b * T_ + t) * H_ + h];
    float* op = out + ((size_t)(b * T_ + t) * H_ + h) * D_;
#pragma unroll
    for (int dt = 0; dt < 4; ++dt)
      op[dt * 16 + row] = accO[dt][r] * gc;
  }
}

// =====================================================================
// Kernel 3: selected attention over the per-token top-4 raw KV blocks.
// One wave = one (b,t,h). Lanes = 32 in-block positions for scores,
// then 2 output dims each for the AV reduction. Accumulates into out.
// =====================================================================
__global__ __launch_bounds__(256) void sel_attn_kernel(
    const float* __restrict__ q, const float* __restrict__ k,
    const float* __restrict__ v, const float* __restrict__ g_slc,
    const int* __restrict__ topk, float* __restrict__ out) {
  __shared__ float qs_s[8][64];
  __shared__ float ps_s[8][32];

  const int w    = threadIdx.x >> 5;
  const int lane = threadIdx.x & 31;
  const int wid  = blockIdx.x * 8 + w;       // 0..32767
  const int h    = wid & 3;
  const int t    = (wid >> 2) & (T_ - 1);
  const int b    = wid >> 13;                // / (H_*T_)
  const size_t g = (size_t)b * T_ + t;

  const float* qrow = q + (g * H_ + h) * D_;
  qs_s[w][lane]      = qrow[lane];
  qs_s[w][lane + 32] = qrow[lane + 32];
  asm volatile("s_wait_dscnt 0x0" ::: "memory");

  float acc0 = 0.0f, acc1 = 0.0f;
  const int* tp = topk + ((size_t)(b * H_ + h) * T_ + t) * S_;
#pragma unroll
  for (int s = 0; s < S_; ++s) {
    const int blk  = tp[s];
    const int blkc = (blk < 0) ? 0 : blk;
    const int pos  = blkc * BS_ + lane;

    // score for this lane's key position
    const float* krow = k + (((size_t)b * T_ + (size_t)blkc * BS_ + lane) * H_ + h) * D_;
    float sc = 0.0f;
#pragma unroll
    for (int d4 = 0; d4 < 16; ++d4) {
      const f32x4 kv = *(const f32x4*)(krow + d4 * 4);
      sc += qs_s[w][d4 * 4 + 0] * kv[0] + qs_s[w][d4 * 4 + 1] * kv[1] +
            qs_s[w][d4 * 4 + 2] * kv[2] + qs_s[w][d4 * 4 + 3] * kv[3];
    }
    sc *= SCALE;
    const float p   = sc / (1.0f + __expf(-sc));
    const bool  msk = (blk >= 0) && (pos <= t);
    ps_s[w][lane]   = msk ? p : 0.0f;
    asm volatile("s_wait_dscnt 0x0" ::: "memory");

    // AV: each lane owns dims (2*lane, 2*lane+1)
    const float* vblk = v + (((size_t)b * T_ + (size_t)blkc * BS_) * H_ + h) * D_;
#pragma unroll 8
    for (int p2 = 0; p2 < BS_; ++p2) {
      const float pv = ps_s[w][p2];
      const f32x2 vv = *(const f32x2*)(vblk + (size_t)p2 * H_ * D_ + 2 * lane);
      acc0 += pv * vv[0];
      acc1 += pv * vv[1];
    }
    asm volatile("s_wait_dscnt 0x0" ::: "memory");
  }

  const float gs = g_slc[g * H_ + h];
  float* op = out + (g * H_ + h) * D_;
  f32x2 o = *(const f32x2*)(op + 2 * lane);
  o[0] += acc0 * gs;
  o[1] += acc1 * gs;
  *(f32x2*)(op + 2 * lane) = o;
}

// =====================================================================
extern "C" void kernel_launch(void* const* d_in, const int* in_sizes, int n_in,
                              void* d_out, int out_size, void* d_ws, size_t ws_size,
                              hipStream_t stream) {
  (void)in_sizes; (void)n_in; (void)out_size; (void)ws_size;
  const float* q     = (const float*)d_in[0];
  const float* k     = (const float*)d_in[1];
  const float* v     = (const float*)d_in[2];
  const float* g_cmp = (const float*)d_in[3];
  const float* g_slc = (const float*)d_in[4];
  // d_in[5] = x_offsets: uniform ragged offsets (B*T), not needed on device.
  float* out = (float*)d_out;

  // workspace layout (768 KB total):
  //   [0      , 131072) : kc  bf16 [B][H][Nb][D]
  //   [131072 , 262144) : vcT bf16 [B][H][D][Nb]
  //   [262144 , 786432) : topk int32 [B][H][T][S]
  char* ws = (char*)d_ws;
  __bf16* kc   = (__bf16*)(ws);
  __bf16* vcT  = (__bf16*)(ws + 131072);
  int*    topk = (int*)   (ws + 262144);

  pool_kernel    <<<B_ * NB_,                  256, 0, stream>>>(k, v, kc, vcT);
  cmp_attn_kernel<<<(B_ * H_ * (T_ / 16)) / 4, 128, 0, stream>>>(q, g_cmp, kc, vcT, topk, out);
  sel_attn_kernel<<<(B_ * T_ * H_) / 8,        256, 0, stream>>>(q, k, v, g_slc, topk, out);
}
